// AttLayer_40441412059745
// MI455X (gfx1250) — matrix operand order repaired and checked by reading, not compiled, not run on hardware
//
#include <hip/hip_runtime.h>

typedef __attribute__((ext_vector_type(16))) _Float16 v16h;
typedef __attribute__((ext_vector_type(8)))  float    v8f;
typedef __attribute__((ext_vector_type(4)))  unsigned int tdm_u32x4;
typedef __attribute__((ext_vector_type(8)))  int          tdm_i32x8;
typedef __attribute__((ext_vector_type(4)))  int          tdm_i32x4;

#define LSEQ 32768
#define CIN  512
#define CP   256
#define NBLK 512   // LSEQ / 64

// Generic LDS pointer -> wave LDS byte offset (ISA: LDS generic addr[31:0] is the offset)
__device__ inline unsigned lds_off(const void* p) {
  return (unsigned)(unsigned long long)p;
}

// ---- Tensor Data Mover: 2D f16 tile load (tile_d0 elems/row, tile_d1 rows,
// row stride stride0 elems). Rows packed contiguously in LDS (pitch = tile_d0).
// Caller guarantees the whole tile is in bounds. Issue once per block (wave 0).
__device__ inline void tdm_load_2d_f16(unsigned lds_addr, const void* gptr,
                                       unsigned tile_d0, unsigned tile_d1,
                                       unsigned long long stride0) {
  unsigned long long ga = (unsigned long long)gptr;
  tdm_u32x4 g0;
  g0[0] = 1u;                                   // count=1, user descriptor
  g0[1] = lds_addr;                             // lds_addr [63:32]
  g0[2] = (unsigned)(ga & 0xffffffffu);         // global_addr[31:0]
  g0[3] = (unsigned)((ga >> 32) & 0x01ffffffu)  // global_addr[56:32]
        | 0x80000000u;                          // type=2 ("image") bits[127:126]=10b
  tdm_i32x8 g1;
  unsigned td0 = tile_d0, td1 = tile_d1;        // tensor dims == tile dims (in-bounds)
  g1[0] = (int)(1u << 16);                      // data_size=1 -> 2 bytes; wg_mask=0
  g1[1] = (int)((td0 & 0xffffu) << 16);         // tensor_dim0[15:0] @ bits 63:48
  g1[2] = (int)(((td0 >> 16) & 0xffffu) | ((td1 & 0xffffu) << 16));   // td0 hi | td1 lo
  g1[3] = (int)(((td1 >> 16) & 0xffffu) | ((tile_d0 & 0xffffu) << 16)); // td1 hi | tile_dim0
  g1[4] = (int)(tile_d1 & 0xffffu);             // tile_dim1 | tile_dim2=0
  g1[5] = (int)(stride0 & 0xffffffffu);         // tensor_dim0_stride[31:0]
  g1[6] = (int)((stride0 >> 32) & 0xffffu);     // stride0[47:32] | stride1[15:0]=0
  g1[7] = 0;
  tdm_i32x4 z4 = {0, 0, 0, 0};
#if defined(__clang_major__) && (__clang_major__ >= 23)
  tdm_i32x8 z8 = {0, 0, 0, 0, 0, 0, 0, 0};
  __builtin_amdgcn_tensor_load_to_lds(g0, g1, z4, z4, z8, 0);
#else
  __builtin_amdgcn_tensor_load_to_lds(g0, g1, z4, z4, 0);
#endif
}

// ---- WMMA fragment loaders (wave32 layouts per CDNA5 ISA 7.12.2) ----
__device__ inline v16h lds_load_a(const _Float16* p0, int ld) {
  int lane = threadIdx.x & 31;
  int row = lane & 15;
  int k0 = (lane & 16) ? 8 : 0;
  const _Float16* p = p0 + row * ld + k0;
  v16h a;
#pragma unroll
  for (int i = 0; i < 8; ++i) { a[i] = p[i]; a[i + 8] = p[i + 16]; }
  return a;
}

__device__ inline v16h lds_load_b(const _Float16* p0, int ld) {
  int lane = threadIdx.x & 31;
  int n = lane & 15;
  int k0 = (lane & 16) ? 8 : 0;
  const _Float16* p = p0 + n * ld + k0;
  v16h b;
#pragma unroll
  for (int i = 0; i < 8; ++i) { b[i] = p[i]; b[i + 8] = p[i + 16]; }
  return b;
}

// ---- Projection GEMM: Wp[CP][CIN] @ X[CIN][L] + b -> f16
// tmode=1: Y stored transposed [L][CP] (for Q/K -> contiguous-K attention tiles)
// tmode=0: Y stored [CP][L]            (for V -> A-operand of att@v)
__global__ __launch_bounds__(256) void proj_kernel(
    const float* __restrict__ W, const float* __restrict__ bias,
    const float* __restrict__ X, _Float16* __restrict__ Y, int tmode) {
  __shared__ _Float16 As[64 * 32];    // [m][k]
  __shared__ _Float16 Bs[128 * 32];   // [n][k]
  const int bn = blockIdx.x * 128;
  const int bm = blockIdx.y * 64;
  const int t = threadIdx.x;
  const int w = t >> 5;
  const int mt = (w & 3) * 16;
  const int ng = (w >> 2) * 64;
  v8f acc[4] = {};
  for (int k0 = 0; k0 < CIN; k0 += 32) {
    __syncthreads();
    {   // stage W tile f32->f16, one 16B LDS store per thread
      int m = t >> 2, ks = (t & 3) * 8;
      const float* src = W + (size_t)(bm + m) * CIN + k0 + ks;
      __align__(16) _Float16 ta[8];
#pragma unroll
      for (int i = 0; i < 8; ++i) ta[i] = (_Float16)src[i];
      *(uint4*)(As + m * 32 + ks) = *(const uint4*)ta;
    }
    {   // stage X tile transposed [n][k]; pack k-pairs -> b32 LDS stores
      int kp = (t >> 4) * 2, n0 = (t & 15) * 8;
      const float* s0 = X + (size_t)(k0 + kp) * LSEQ + bn + n0;
      const float* s1 = s0 + LSEQ;
#pragma unroll
      for (int i = 0; i < 8; ++i) {
        union { unsigned u; _Float16 h[2]; } pk;
        pk.h[0] = (_Float16)s0[i];
        pk.h[1] = (_Float16)s1[i];
        *(unsigned*)(Bs + (n0 + i) * 32 + kp) = pk.u;
      }
    }
    __syncthreads();
    v16h a = lds_load_a(As + mt * 32, 32);
#pragma unroll
    for (int j = 0; j < 4; ++j) {
      v16h b = lds_load_b(Bs + (ng + j * 16) * 32, 32);
      acc[j] = __builtin_amdgcn_wmma_f32_16x16x32_f16(false, a, false, b,
                                                      (short)0, acc[j], false, false);
    }
  }
  const int lane = t & 31, hi = (lane >> 4) * 8, nn = lane & 15;
  if (tmode) {
#pragma unroll
    for (int j = 0; j < 4; ++j) {
      __align__(16) _Float16 tmp[8];
#pragma unroll
      for (int r = 0; r < 8; ++r)
        tmp[r] = (_Float16)(acc[j][r] + bias[bm + mt + hi + r]);
      int n = bn + ng + j * 16 + nn;
      *(uint4*)(Y + (size_t)n * CP + bm + mt + hi) = *(const uint4*)tmp;
    }
  } else {
#pragma unroll
    for (int j = 0; j < 4; ++j)
#pragma unroll
      for (int r = 0; r < 8; ++r) {
        int m = bm + mt + hi + r;
        int n = bn + ng + j * 16 + nn;
        Y[(size_t)m * LSEQ + n] = (_Float16)(acc[j][r] + bias[m]);
      }
  }
}

// ---- Output GEMM: out[CIN][L] = (Wo[CIN][CP] @ R^T + bo) * mask ; R stored [L][CP]
__global__ __launch_bounds__(256) void outproj_kernel(
    const float* __restrict__ W, const float* __restrict__ bias,
    const _Float16* __restrict__ Rt, const float* __restrict__ maskp,
    float* __restrict__ Y) {
  __shared__ _Float16 As[64 * 32];
  __shared__ _Float16 Bs[128 * 32];   // [n][k], filled by TDM (rows contiguous in R^T)
  const int bn = blockIdx.x * 128;
  const int bm = blockIdx.y * 64;
  const int t = threadIdx.x;
  const int w = t >> 5;
  const int mt = (w & 3) * 16;
  const int ng = (w >> 2) * 64;
  v8f acc[4] = {};
  for (int k0 = 0; k0 < CP; k0 += 32) {
    __syncthreads();
    {
      int m = t >> 2, ks = (t & 3) * 8;
      const float* src = W + (size_t)(bm + m) * CP + k0 + ks;
      __align__(16) _Float16 ta[8];
#pragma unroll
      for (int i = 0; i < 8; ++i) ta[i] = (_Float16)src[i];
      *(uint4*)(As + m * 32 + ks) = *(const uint4*)ta;
    }
    if (w == 0) {   // TDM: 128 rows x 32 halfs, row stride CP; always in-bounds
      tdm_load_2d_f16(lds_off(Bs), Rt + (size_t)bn * CP + k0, 32, 128, CP);
      __builtin_amdgcn_s_wait_tensorcnt(0);
    }
    __syncthreads();
    v16h a = lds_load_a(As + mt * 32, 32);
#pragma unroll
    for (int j = 0; j < 4; ++j) {
      v16h b = lds_load_b(Bs + (ng + j * 16) * 32, 32);
      acc[j] = __builtin_amdgcn_wmma_f32_16x16x32_f16(false, a, false, b,
                                                      (short)0, acc[j], false, false);
    }
  }
  const int lane = t & 31, hi = (lane >> 4) * 8, nn = lane & 15;
#pragma unroll
  for (int j = 0; j < 4; ++j)
#pragma unroll
    for (int r = 0; r < 8; ++r) {
      int m = bm + mt + hi + r;
      int n = bn + ng + j * 16 + nn;
      Y[(size_t)m * LSEQ + n] = (acc[j][r] + bias[m]) * maskp[n];
    }
}

// ---- Windowed attention per 64-block: energy -> softmax -> att@v -> relu ----
// Qt/Kt stored [L][CP] (K contiguous), Vf stored [CP][L], R out [L][CP].
__global__ __launch_bounds__(256) void attn_kernel(
    const _Float16* __restrict__ Qt, const _Float16* __restrict__ Kt,
    const _Float16* __restrict__ Vf, const float* __restrict__ maskp,
    _Float16* __restrict__ R) {
  // LDS (aliased by phase):
  //  [0,4096):   Qs 64x32 f16      (phase 1)
  //  [4096,12288): Ks 128x32 f16   (phase 1)
  //  [0,33792):  Es 64x132 f32     (phase 2)
  //  [0,16384):  Vs 256x32 f16     (phase 3)
  //  [33792,51200): att 64x136 f16
  __shared__ __align__(16) unsigned char smem[51200];
  _Float16* Qs  = (_Float16*)smem;
  _Float16* Ks  = (_Float16*)(smem + 4096);
  float*    Es  = (float*)smem;
  _Float16* Vs  = (_Float16*)smem;
  _Float16* att = (_Float16*)(smem + 33792);

  const int base = (int)blockIdx.x * 64;
  const bool interior = (blockIdx.x > 0) && (blockIdx.x < NBLK - 1);
  const int t = threadIdx.x;
  const int w = t >> 5;
  const int lane = t & 31;

  // Phase 1: energy[q][p] = sum_c qb[q][c]*kb[p][c]  (M=64, N=128, K=256)
  const int mt = (w & 3) * 16;
  const int ng = (w >> 2) * 64;
  v8f eacc[4] = {};
  for (int c0 = 0; c0 < CP; c0 += 32) {
    __syncthreads();
    if (interior) {
      if (w == 0) {
        tdm_load_2d_f16(lds_off(Qs), Qt + (size_t)base * CP + c0, 32, 64, CP);
        tdm_load_2d_f16(lds_off(Ks), Kt + (size_t)(base - 32) * CP + c0, 32, 128, CP);
        __builtin_amdgcn_s_wait_tensorcnt(0);
      }
    } else {
      {   // Qs[q][c0..c0+32): contiguous rows
        int qq = t >> 2, cs = (t & 3) * 8;
        *(uint4*)(Qs + qq * 32 + cs) =
            *(const uint4*)(Qt + (size_t)(base + qq) * CP + c0 + cs);
      }
      {   // Ks[p][c0..): zero-pad OOB window rows
        int p = t >> 1, cs = (t & 1) * 16;
        int pos = base - 32 + p;
        uint4 va = {0, 0, 0, 0}, vb = {0, 0, 0, 0};
        if (pos >= 0 && pos < LSEQ) {
          const uint4* s = (const uint4*)(Kt + (size_t)pos * CP + c0 + cs);
          va = s[0]; vb = s[1];
        }
        uint4* d = (uint4*)(Ks + p * 32 + cs);
        d[0] = va; d[1] = vb;
      }
    }
    __syncthreads();
    v16h a = lds_load_a(Qs + mt * 32, 32);
#pragma unroll
    for (int j = 0; j < 4; ++j) {
      v16h b = lds_load_b(Ks + (ng + j * 16) * 32, 32);
      eacc[j] = __builtin_amdgcn_wmma_f32_16x16x32_f16(false, a, false, b,
                                                       (short)0, eacc[j], false, false);
    }
  }
  __syncthreads();
  {   // dump scaled energy (1/sqrt(256) = 1/16)
    int hi = (lane >> 4) * 8, nn = lane & 15;
#pragma unroll
    for (int j = 0; j < 4; ++j)
#pragma unroll
      for (int r = 0; r < 8; ++r)
        Es[(mt + hi + r) * 132 + (ng + j * 16 + nn)] = eacc[j][r] * 0.0625f;
  }
  __syncthreads();
  // Phase 2: softmax(energy + log(fm+1e-6)) * fm (masked terms stay in denominator)
  if (t < 64) {
    const int q = t;
    float mx = -1e30f;
    for (int j = 0; j < 128; ++j) {
      int pos = base - 32 + j;
      float pm = (pos >= 0 && pos < LSEQ) ? maskp[pos] : 0.0f;
      float fm = ((j >= q) && (j < q + 64)) ? pm : 0.0f;
      mx = fmaxf(mx, Es[q * 132 + j] + __logf(fm + 1e-6f));
    }
    float sum = 0.0f;
    for (int j = 0; j < 128; ++j) {
      int pos = base - 32 + j;
      float pm = (pos >= 0 && pos < LSEQ) ? maskp[pos] : 0.0f;
      float fm = ((j >= q) && (j < q + 64)) ? pm : 0.0f;
      float p = __expf(Es[q * 132 + j] + __logf(fm + 1e-6f) - mx);
      sum += p;
      att[q * 136 + j] = (_Float16)(p * fm);
    }
    float inv = 1.0f / sum;
    for (int j = 0; j < 128; ++j)
      att[q * 136 + j] = (_Float16)((float)att[q * 136 + j] * inv);
  }
  __syncthreads();
  // Phase 3: out[c][q] = sum_p vb[c][p]*att[q][p]  (M=256, N=64, K=128)
  const int m0 = w * 32;
  v8f oacc[2][4] = {};
  for (int kk0 = 0; kk0 < 128; kk0 += 32) {
    if (interior) {
      if (w == 0) {   // 256 rows x 32 halfs, row stride LSEQ
        tdm_load_2d_f16(lds_off(Vs), Vf + (size_t)(base - 32 + kk0), 32, 256, LSEQ);
        __builtin_amdgcn_s_wait_tensorcnt(0);
      }
    } else {
      const _Float16* src = Vf + (size_t)t * LSEQ;
#pragma unroll
      for (int i = 0; i < 32; ++i) {
        int pos = base - 32 + kk0 + i;
        _Float16 vv = (_Float16)0.0f;
        if (pos >= 0 && pos < LSEQ) vv = src[pos];
        Vs[t * 32 + i] = vv;
      }
    }
    __syncthreads();
    v16h a0 = lds_load_a(Vs + m0 * 32, 32);
    v16h a1 = lds_load_a(Vs + (m0 + 16) * 32, 32);
#pragma unroll
    for (int j = 0; j < 4; ++j) {
      v16h b = lds_load_b(att + (j * 16) * 136 + kk0, 136);
      oacc[0][j] = __builtin_amdgcn_wmma_f32_16x16x32_f16(false, a0, false, b,
                                                          (short)0, oacc[0][j], false, false);
      oacc[1][j] = __builtin_amdgcn_wmma_f32_16x16x32_f16(false, a1, false, b,
                                                          (short)0, oacc[1][j], false, false);
    }
    __syncthreads();
  }
  {   // relu -> R stored [L][CP]: one 16B store per tile per lane
    int hi = (lane >> 4) * 8, nn = lane & 15;
#pragma unroll
    for (int mi = 0; mi < 2; ++mi)
#pragma unroll
      for (int j = 0; j < 4; ++j) {
        __align__(16) _Float16 tmp[8];
#pragma unroll
        for (int r = 0; r < 8; ++r) tmp[r] = (_Float16)fmaxf(oacc[mi][j][r], 0.0f);
        int qq = j * 16 + nn;
        *(uint4*)(R + (size_t)(base + qq) * CP + m0 + mi * 16 + hi) = *(const uint4*)tmp;
      }
  }
}

extern "C" void kernel_launch(void* const* d_in, const int* in_sizes, int n_in,
                              void* d_out, int out_size, void* d_ws, size_t ws_size,
                              hipStream_t stream) {
  (void)in_sizes; (void)n_in; (void)out_size; (void)ws_size;
  const float* q    = (const float*)d_in[0];
  const float* k    = (const float*)d_in[1];
  const float* v    = (const float*)d_in[2];
  const float* mask = (const float*)d_in[3];
  const float* Wq   = (const float*)d_in[4];
  const float* bq   = (const float*)d_in[5];
  const float* Wk   = (const float*)d_in[6];
  const float* bk   = (const float*)d_in[7];
  const float* Wv   = (const float*)d_in[8];
  const float* bv   = (const float*)d_in[9];
  const float* Wo   = (const float*)d_in[10];
  const float* bo   = (const float*)d_in[11];
  float* out = (float*)d_out;

  const size_t plane = (size_t)CP * LSEQ;   // f16 elements per projected tensor
  _Float16* Qt = (_Float16*)d_ws;           // [L][CP]
  _Float16* Kt = Qt + plane;                // [L][CP]
  _Float16* Vf = Kt + plane;                // [CP][L]
  _Float16* Rt = Vf + plane;                // [L][CP]  (needs 64 MB of d_ws)

  dim3 blk(256);
  dim3 gproj(LSEQ / 128, CP / 64);
  proj_kernel<<<gproj, blk, 0, stream>>>(Wq, bq, q, Qt, 1);
  proj_kernel<<<gproj, blk, 0, stream>>>(Wk, bk, k, Kt, 1);
  proj_kernel<<<gproj, blk, 0, stream>>>(Wv, bv, v, Vf, 0);
  attn_kernel<<<dim3(NBLK), blk, 0, stream>>>(Qt, Kt, Vf, mask, Rt);
  outproj_kernel<<<dim3(LSEQ / 128, CIN / 64), blk, 0, stream>>>(Wo, bo, Rt, mask, out);
}
